// EMLLABlock_36232344109197
// MI455X (gfx1250) — matrix operand order repaired and checked by reading, not compiled, not run on hardware
//
#include <hip/hip_runtime.h>
#include <hip/hip_bf16.h>
#include <math.h>
#include <stdint.h>

typedef __bf16 bf16;
typedef __attribute__((ext_vector_type(16))) bf16 bf16x16;
typedef __attribute__((ext_vector_type(8)))  bf16 bf16x8;
typedef __attribute__((ext_vector_type(8)))  float v8f;

#define MTOT  50176   // B * L
#define CDIM  192
#define LTOK  3136    // 56*56
#define BIMG  16
#define NHEAD 6
#define HDIM  32

__device__ __forceinline__ float siluf(float x){ return x / (1.0f + __expf(-x)); }
__device__ __forceinline__ float elup1(float x){ return x > 0.0f ? x + 1.0f : __expf(x); }

// LDS byte offset of a __shared__ pointer (ptrtoint of an addrspace(3) ptr).
typedef __attribute__((address_space(3))) void lds_void;
__device__ __forceinline__ unsigned lds_off(void* p){
  return (unsigned)(unsigned long long)(lds_void*)p;
}

// ---------------------------------------------------------------------------
// WMMA fragment helpers (v_wmma_f32_16x16x32_bf16, wave32)
// A (16x32, 16-bit) ISA layout: lanes 0-15 hold row M=lane with K pairs
// {0..7,16..23}; lanes 16-31 hold row M=lane-16 with K pairs {8..15,24..31}.
__device__ __forceinline__ bf16x16 load_a_frag(const bf16* base, int row0,
                                               int ld, int k0, int lane)
{
  const int m  = lane & 15;
  const int ko = (lane & 16) ? 8 : 0;
  const bf16* p = base + (size_t)(row0 + m) * ld + (k0 + ko);
  bf16x8 lo = *(const bf16x8*)(p);        // K = k0+ko   .. +7
  bf16x8 hi = *(const bf16x8*)(p + 16);   // K = k0+16+ko.. +7
  bf16x16 a;
#pragma unroll
  for (int i = 0; i < 8; ++i){ a[i] = lo[i]; a[i + 8] = hi[i]; }
  return a;
}

// B (32x16): lanes 0-15 hold column N=lane, K {0..7,16..23}; lanes 16-31 hold
// column N=lane-16, K {8..15,24..31}.  Weights staged TRANSPOSED in LDS.
__device__ __forceinline__ bf16x16 load_b_frag_t(const bf16* wt, int n0,
                                                 int ldk, int k0, int lane)
{
  const int n   = n0 + (lane & 15);
  const int koB = (lane & 16) ? 8 : 0;
  const bf16* p = wt + (size_t)n * ldk + (k0 + koB);
  bf16x8 lo = *(const bf16x8*)(p);
  bf16x8 hi = *(const bf16x8*)(p + 16);
  bf16x16 b;
#pragma unroll
  for (int i = 0; i < 8; ++i){ b[i] = lo[i]; b[8 + i] = hi[i]; }
  return b;
}

#define WMMA_BF16(A, B, C) \
  __builtin_amdgcn_wmma_f32_16x16x32_bf16(false, (A), false, (B), (short)0, (C), false, false)

// ---------------------------------------------------------------------------
// 1) LayerNorm: one wave per row (C=192 -> 6 elems/lane), bf16 output.
__global__ void __launch_bounds__(256)
ln_kernel(const float* __restrict__ x, const float* __restrict__ g,
          const float* __restrict__ bta, bf16* __restrict__ xn)
{
  const int w = threadIdx.x >> 5, lane = threadIdx.x & 31;
  const int row = blockIdx.x * 8 + w;
  const float* xr = x + (size_t)row * CDIM;
  float v[6]; float s = 0.f, sq = 0.f;
#pragma unroll
  for (int j = 0; j < 6; ++j){ v[j] = xr[lane + 32*j]; s += v[j]; sq += v[j]*v[j]; }
#pragma unroll
  for (int off = 16; off >= 1; off >>= 1){
    s  += __shfl_xor(s,  off, 32);
    sq += __shfl_xor(sq, off, 32);
  }
  const float mu  = s * (1.0f/192.0f);
  const float var = sq * (1.0f/192.0f) - mu*mu;
  const float inv = rsqrtf(var + 1e-5f);
  bf16* xo = xn + (size_t)row * CDIM;
#pragma unroll
  for (int j = 0; j < 6; ++j){
    const int c = lane + 32*j;
    xo[c] = (bf16)((v[j] - mu) * inv * g[c] + bta[c]);
  }
}

// ---------------------------------------------------------------------------
// 2) Fused dual GEMM: xn@act_w(+b,silu)->act_res(f32), xn@in_w(+b)->hpre(f32)
__global__ void __launch_bounds__(128)
gemm_dual_kernel(const bf16* __restrict__ xn,
                 const float* __restrict__ w1, const float* __restrict__ b1,
                 const float* __restrict__ w2, const float* __restrict__ b2,
                 float* __restrict__ out1, float* __restrict__ out2)
{
  extern __shared__ char smem_raw[];
  bf16* wt1 = (bf16*)smem_raw;              // [192][192] transposed
  bf16* wt2 = wt1 + CDIM * CDIM;
  const int tid = threadIdx.x;
  for (int i = tid; i < CDIM*CDIM; i += 128){
    const int n = i / CDIM, k = i - n * CDIM;
    wt1[i] = (bf16)w1[(size_t)k * CDIM + n];
    wt2[i] = (bf16)w2[(size_t)k * CDIM + n];
  }
  __syncthreads();
  const int w = tid >> 5, lane = tid & 31;
  const int row0 = blockIdx.x * 64 + w * 16;
  bf16x16 af[6];
#pragma unroll
  for (int kc = 0; kc < 6; ++kc) af[kc] = load_a_frag(xn, row0, CDIM, kc*32, lane);
  const int rsub = (lane & 16) ? 8 : 0;
  const int nloc = lane & 15;
  for (int nt = 0; nt < 12; ++nt){
    v8f acc1 = {}, acc2 = {};
#pragma unroll
    for (int kc = 0; kc < 6; ++kc){
      bf16x16 bf1 = load_b_frag_t(wt1, nt*16, CDIM, kc*32, lane);
      acc1 = WMMA_BF16(af[kc], bf1, acc1);
      bf16x16 bf2 = load_b_frag_t(wt2, nt*16, CDIM, kc*32, lane);
      acc2 = WMMA_BF16(af[kc], bf2, acc2);
    }
    const int col = nt*16 + nloc;
    const float bias1 = b1[col], bias2 = b2[col];
#pragma unroll
    for (int r = 0; r < 8; ++r){
      const int row = row0 + r + rsub;
      out1[(size_t)row*CDIM + col] = siluf(acc1[r] + bias1);
      out2[(size_t)row*CDIM + col] = acc2[r] + bias2;
    }
  }
}

// ---------------------------------------------------------------------------
// 3/5) Depthwise 3x3 conv (SAME, zero pad), optional silu, optional bf16 copy
__global__ void __launch_bounds__(256)
dwconv_kernel(const float* __restrict__ in, const float* __restrict__ wt,
              const float* __restrict__ bias, float* __restrict__ out32,
              bf16* __restrict__ out16, int apply_silu)
{
  const int idx = blockIdx.x * 256 + threadIdx.x;
  const int c = idx % CDIM;
  int t = idx / CDIM;
  const int px = t % 56; t /= 56;
  const int py = t % 56; const int b = t / 56;
  float acc = bias[c];
#pragma unroll
  for (int ky = 0; ky < 3; ++ky){
    const int yy = py + ky - 1;
    if (yy < 0 || yy >= 56) continue;
#pragma unroll
    for (int kx = 0; kx < 3; ++kx){
      const int xx = px + kx - 1;
      if (xx < 0 || xx >= 56) continue;
      acc += in[(((size_t)(b*56 + yy))*56 + xx)*CDIM + c] * wt[(ky*3 + kx)*CDIM + c];
    }
  }
  if (apply_silu) acc = siluf(acc);
  out32[idx] = acc;
  if (out16) out16[idx] = (bf16)acc;
}

// ---------------------------------------------------------------------------
// 4) GEMM h@qk_w (N=384) + bias, elu+1, split into q/k bf16 buffers
__global__ void __launch_bounds__(128)
gemm_qk_kernel(const bf16* __restrict__ h16, const float* __restrict__ wqk,
               const float* __restrict__ bqk, bf16* __restrict__ qout,
               bf16* __restrict__ kout)
{
  extern __shared__ char smem_raw[];
  bf16* wt = (bf16*)smem_raw;               // [384][192] transposed
  const int tid = threadIdx.x;
  for (int i = tid; i < 384*CDIM; i += 128){
    const int n = i / CDIM, k = i - n * CDIM;
    wt[i] = (bf16)wqk[(size_t)k * 384 + n];
  }
  __syncthreads();
  const int w = tid >> 5, lane = tid & 31;
  const int row0 = blockIdx.x * 64 + w * 16;
  bf16x16 af[6];
#pragma unroll
  for (int kc = 0; kc < 6; ++kc) af[kc] = load_a_frag(h16, row0, CDIM, kc*32, lane);
  const int rsub = (lane & 16) ? 8 : 0;
  const int nloc = lane & 15;
  for (int nt = 0; nt < 24; ++nt){
    v8f acc = {};
#pragma unroll
    for (int kc = 0; kc < 6; ++kc){
      bf16x16 bfr = load_b_frag_t(wt, nt*16, CDIM, kc*32, lane);
      acc = WMMA_BF16(af[kc], bfr, acc);
    }
    const int col = nt*16 + nloc;
    const float bias = bqk[col];
    bf16* dst = (col < CDIM) ? qout : kout;
    const int ccol = (col < CDIM) ? col : col - CDIM;
#pragma unroll
    for (int r = 0; r < 8; ++r){
      const int row = row0 + r + rsub;
      dst[(size_t)row*CDIM + ccol] = (bf16)elup1(acc[r] + bias);
    }
  }
}

// ---------------------------------------------------------------------------
// 6) Per-(b,head): kv = (k^T v)/L (32x32) via WMMA K-reduction over L=3136,
//    plus k_mean.  8 waves: 2x2 output tiles x 2 K-halves per 64-row chunk.
//    Chunks are double-buffered through LDS with GLOBAL_LOAD_ASYNC_TO_LDS,
//    synchronized with s_wait_asynccnt (async loads complete in order).
__global__ void __launch_bounds__(256)
kv_kernel(const bf16* __restrict__ kbuf, const bf16* __restrict__ vbuf,
          bf16* __restrict__ kv, float* __restrict__ kmean)
{
  __shared__ __align__(16) bf16 kc[2][64][32];
  __shared__ __align__(16) bf16 vc[2][64][32];
  __shared__ float red[8][8][32];
  __shared__ float ksr[8][32];
  const int bh = blockIdx.x;
  const int b = bh / NHEAD, hh = bh % NHEAD;
  const int tid = threadIdx.x;
  const int w = tid >> 5, lane = tid & 31;
  const int mt = (w >> 1) & 1, nt = w & 1, half = w >> 2;
  const int lb = half * 32;
  const int m   = mt*16 + (lane & 15);
  const int ko  = (lane & 16) ? 8 : 0;
  const int n   = nt*16 + (lane & 15);
  const int koB = (lane & 16) ? 16 : 0;
  v8f acc = {};
  float ksum = 0.f;

  // each thread owns one 16B transfer per buffer per chunk
  const int lld = tid >> 2, o8 = (tid & 3) * 8;
  const size_t gbase = ((size_t)(b*LTOK + lld))*CDIM + hh*HDIM + o8;
  const unsigned klds[2] = { lds_off(&kc[0][lld][o8]), lds_off(&kc[1][lld][o8]) };
  const unsigned vlds[2] = { lds_off(&vc[0][lld][o8]), lds_off(&vc[1][lld][o8]) };

  #define ISSUE_CHUNK(chunk)                                                   \
    do {                                                                       \
      const int bs_ = (chunk) & 1;                                             \
      const uint64_t ka_ = (uint64_t)(kbuf + gbase + (size_t)(chunk)*64*CDIM); \
      const uint64_t va_ = (uint64_t)(vbuf + gbase + (size_t)(chunk)*64*CDIM); \
      asm volatile("global_load_async_to_lds_b128 %0, %1, off"                 \
                   :: "v"(klds[bs_]), "v"(ka_) : "memory");                    \
      asm volatile("global_load_async_to_lds_b128 %0, %1, off"                 \
                   :: "v"(vlds[bs_]), "v"(va_) : "memory");                    \
    } while (0)

  const int NCHUNK = LTOK / 64;   // 49
  ISSUE_CHUNK(0);
  for (int c = 0; c < NCHUNK; ++c){
    const int cur = c & 1;
    if (c + 1 < NCHUNK){
      ISSUE_CHUNK(c + 1);                       // prefetch into other buffer
      asm volatile("s_wait_asynccnt 0x2" ::: "memory");  // chunk c landed
    } else {
      asm volatile("s_wait_asynccnt 0x0" ::: "memory");
    }
    __syncthreads();                            // all waves' chunk-c data visible
#pragma unroll
    for (int j = 0; j < 8; ++j) ksum += (float)kc[cur][w + 8*j][lane];
    bf16x16 a, bb;
#pragma unroll
    for (int e = 0; e < 16; ++e){
      const int kk = e + ko + ((e >= 8) ? 8 : 0);   // A-layout K index
      a[e]  = kc[cur][lb + kk][m];                   // transposed gather: k^T
      bb[e] = vc[cur][lb + kk][n];
    }
    acc = WMMA_BF16(a, bb, acc);
    __syncthreads();                            // buffer free before re-fill
  }
  #undef ISSUE_CHUNK

  ksr[w][lane] = ksum;
#pragma unroll
  for (int r = 0; r < 8; ++r) red[w][r][lane] = acc[r];
  __syncthreads();
  if (w == 0){
    float t = 0.f;
#pragma unroll
    for (int j = 0; j < 8; ++j) t += ksr[j][lane];
    kmean[bh*HDIM + lane] = t * (1.0f/(float)LTOK);
  }
  if (w < 4){
    const float scale = 1.0f/(float)LTOK;   // inv_sqrt_n^2
    const int rsub = (lane & 16) ? 8 : 0;
#pragma unroll
    for (int r = 0; r < 8; ++r){
      const float sv = (red[w][r][lane] + red[w + 4][r][lane]) * scale;
      const int mr = mt*16 + r + rsub;
      const int nc = nt*16 + (lane & 15);
      kv[(size_t)bh*1024 + mr*32 + nc] = (bf16)sv;
    }
  }
}

// ---------------------------------------------------------------------------
// 7) att = (q @ kv) * z + lepe;  m = att * act_res (bf16).
__global__ void __launch_bounds__(256)
att_kernel(const bf16* __restrict__ q, const bf16* __restrict__ kv,
           const float* __restrict__ kmean, const float* __restrict__ lepe,
           const float* __restrict__ actres, bf16* __restrict__ mout)
{
  const int tid = threadIdx.x;
  const int w = tid >> 5, lane = tid & 31;
  const int row0 = blockIdx.x * 128 + w * 16;  // 16-row tiles never cross batch
  const int hh = blockIdx.y;
  const int b = row0 / LTOK;
  const int bh = b * NHEAD + hh;
  const bf16* kvb = kv + (size_t)bh * 1024;    // [32][32] k-major
  const int koB = (lane & 16) ? 16 : 0;
  const int nloc = lane & 15;
  bf16x16 b0, b1;
#pragma unroll
  for (int e = 0; e < 16; ++e){
    const int kk = (koB >> 1) + e + ((e >= 8) ? 8 : 0);
    b0[e] = kvb[kk*32 + nloc];
    b1[e] = kvb[kk*32 + 16 + nloc];
  }
  const bf16x16 a = load_a_frag(q, row0, CDIM, hh*HDIM, lane);
  v8f acc0 = {}, acc1 = {};
  acc0 = WMMA_BF16(a, b0, acc0);
  acc1 = WMMA_BF16(a, b1, acc1);
  const int zrow = row0 + (lane & 15);
  const bf16* qr = q + (size_t)zrow * CDIM + hh*HDIM;
  const float* km = kmean + bh*HDIM;
  float dot = 0.f;
#pragma unroll
  for (int d = 0; d < HDIM; ++d) dot += (float)qr[d] * km[d];
  const float z = 1.0f / (dot + 1e-6f);
  const int rsub = (lane & 16) ? 8 : 0;
#pragma unroll
  for (int r = 0; r < 8; ++r){
    const int row = row0 + r + rsub;
    const float zr = __shfl(z, r + rsub, 32);
    {
      const int c = hh*HDIM + nloc;
      const size_t off = (size_t)row*CDIM + c;
      const float att = acc0[r] * zr + lepe[off];
      mout[off] = (bf16)(att * actres[off]);
    }
    {
      const int c = hh*HDIM + 16 + nloc;
      const size_t off = (size_t)row*CDIM + c;
      const float att = acc1[r] * zr + lepe[off];
      mout[off] = (bf16)(att * actres[off]);
    }
  }
}

// ---------------------------------------------------------------------------
// 8) out = m @ out_w + out_b + shortcut(x)   (f32 output)
__global__ void __launch_bounds__(128)
gemm_out_kernel(const bf16* __restrict__ m, const float* __restrict__ wo,
                const float* __restrict__ bo, const float* __restrict__ x,
                float* __restrict__ out)
{
  extern __shared__ char smem_raw[];
  bf16* wt = (bf16*)smem_raw;               // [192][192] transposed
  const int tid = threadIdx.x;
  for (int i = tid; i < CDIM*CDIM; i += 128){
    const int n = i / CDIM, k = i - n * CDIM;
    wt[i] = (bf16)wo[(size_t)k * CDIM + n];
  }
  __syncthreads();
  const int w = tid >> 5, lane = tid & 31;
  const int row0 = blockIdx.x * 64 + w * 16;
  bf16x16 af[6];
#pragma unroll
  for (int kc = 0; kc < 6; ++kc) af[kc] = load_a_frag(m, row0, CDIM, kc*32, lane);
  const int rsub = (lane & 16) ? 8 : 0;
  const int nloc = lane & 15;
  for (int nt = 0; nt < 12; ++nt){
    v8f acc = {};
#pragma unroll
    for (int kc = 0; kc < 6; ++kc){
      bf16x16 bfr = load_b_frag_t(wt, nt*16, CDIM, kc*32, lane);
      acc = WMMA_BF16(af[kc], bfr, acc);
    }
    const int col = nt*16 + nloc;
    const float bias = bo[col];
#pragma unroll
    for (int r = 0; r < 8; ++r){
      const int row = row0 + r + rsub;
      const size_t off = (size_t)row*CDIM + col;
      out[off] = acc[r] + bias + x[off];
    }
  }
}

// ---------------------------------------------------------------------------
extern "C" void kernel_launch(void* const* d_in, const int* in_sizes, int n_in,
                              void* d_out, int out_size, void* d_ws, size_t ws_size,
                              hipStream_t stream)
{
  const float* x      = (const float*)d_in[0];
  const float* norm_g = (const float*)d_in[1];
  const float* norm_b = (const float*)d_in[2];
  const float* in_w   = (const float*)d_in[3];
  const float* in_b   = (const float*)d_in[4];
  const float* act_w  = (const float*)d_in[5];
  const float* act_b  = (const float*)d_in[6];
  const float* dwc_w  = (const float*)d_in[7];
  const float* dwc_b  = (const float*)d_in[8];
  const float* qk_w   = (const float*)d_in[9];
  const float* qk_b   = (const float*)d_in[10];
  const float* lepe_w = (const float*)d_in[11];
  const float* lepe_b = (const float*)d_in[12];
  const float* out_w  = (const float*)d_in[13];
  const float* out_b  = (const float*)d_in[14];

  char* ws = (char*)d_ws;
  const size_t MC16 = (size_t)MTOT * CDIM * 2;   // bf16 plane
  const size_t MC32 = (size_t)MTOT * CDIM * 4;   // f32 plane
  bf16*  xn_m = (bf16*)(ws);                          // xn, reused as m
  float* act  = (float*)(ws + MC16);                  // act_res (f32)
  float* hpre = (float*)(ws + MC16 + 1*MC32);         // hpre, reused as lepe
  float* h32  = (float*)(ws + MC16 + 2*MC32);         // h (f32)
  bf16*  h16  = (bf16*)(ws + MC16 + 3*MC32);          // h (bf16) == v
  bf16*  qb   = (bf16*)(ws + 2*MC16 + 3*MC32);        // q (bf16)
  bf16*  kb   = (bf16*)(ws + 3*MC16 + 3*MC32);        // k (bf16)
  bf16*  kv   = (bf16*)(ws + 4*MC16 + 3*MC32);        // kv (bf16, 96*1024)
  float* km   = (float*)(ws + 4*MC16 + 3*MC32 +
                         (size_t)BIMG*NHEAD*HDIM*HDIM*2); // k_mean (f32)

  ln_kernel<<<MTOT/8, 256, 0, stream>>>(x, norm_g, norm_b, xn_m);
  gemm_dual_kernel<<<MTOT/64, 128, 2*CDIM*CDIM*sizeof(bf16), stream>>>(
      xn_m, act_w, act_b, in_w, in_b, act, hpre);
  dwconv_kernel<<<(MTOT*CDIM)/256, 256, 0, stream>>>(
      hpre, dwc_w, dwc_b, h32, h16, 1);
  gemm_qk_kernel<<<MTOT/64, 128, 2*CDIM*CDIM*sizeof(bf16), stream>>>(
      h16, qk_w, qk_b, qb, kb);
  dwconv_kernel<<<(MTOT*CDIM)/256, 256, 0, stream>>>(
      h32, lepe_w, lepe_b, hpre /*lepe*/, nullptr, 0);
  kv_kernel<<<BIMG*NHEAD, 256, 0, stream>>>(kb, h16, kv, km);
  att_kernel<<<dim3(MTOT/128, NHEAD), 256, 0, stream>>>(
      qb, kv, km, hpre /*lepe*/, act, xn_m /*m*/);
  gemm_out_kernel<<<MTOT/64, 128, CDIM*CDIM*sizeof(bf16), stream>>>(
      xn_m, out_w, out_b, x, (float*)d_out);
}
